// SparseAttention_15908558865360
// MI455X (gfx1250) — matrix operand order repaired and checked by reading, not compiled, not run on hardware
//
#include <hip/hip_runtime.h>

// Problem constants (from reference)
#define B_     2
#define L_     4096
#define D_     1024
#define H_     16
#define DH     64
#define TOPK_  64
#define W_     128
#define NC_    32
#define LW_    256
#define NKEY   320              // LW + TOPK
#define SCALE_ 0.125f           // 64^-0.5
#define QKVS   3072             // fused QKV row stride
#define KOFF_  1024
#define VOFF_  2048

typedef __bf16 bf16_t;
typedef bf16_t v16bf __attribute__((ext_vector_type(16)));
typedef float  v8f   __attribute__((ext_vector_type(8)));

union FragBF {
  v16bf v;
  uint4 q[2];
  unsigned short s[16];
};

__device__ __forceinline__ unsigned short f2bf(float f) {
  unsigned int u = __float_as_uint(f);
  u += 0x7FFFu + ((u >> 16) & 1u);          // round-to-nearest-even
  return (unsigned short)(u >> 16);
}

__device__ __forceinline__ v8f wmma_bf16(v16bf a, v16bf b, v8f c) {
  // v_wmma_f32_16x16x32_bf16: D = A(16x32) * B(32x16) + C(16x16 f32)
  return __builtin_amdgcn_wmma_f32_16x16x32_bf16(false, a, false, b, (short)0, c,
                                                 false, false);
}

// ---------------------------------------------------------------- converts
__global__ __launch_bounds__(256) void k_f32_to_bf16(const float* __restrict__ in,
                                                     unsigned short* __restrict__ out,
                                                     int n) {
  int i = blockIdx.x * 256 + threadIdx.x;
  if (i < n) out[i] = f2bf(in[i]);
}

// Wt[n][k] = bf16(W[k][n])   (W is K x N row-major, K = N = 1024)
__global__ __launch_bounds__(256) void k_transpose_bf16(const float* __restrict__ W,
                                                        unsigned short* __restrict__ Wt) {
  int idx = blockIdx.x * 256 + threadIdx.x;   // over 1M elements
  int n = idx >> 10, k = idx & 1023;
  Wt[n * 1024 + k] = f2bf(W[k * 1024 + n]);
}

__global__ __launch_bounds__(256) void k_concat_bias(const float* __restrict__ bq,
                                                     const float* __restrict__ bk,
                                                     const float* __restrict__ bv,
                                                     float* __restrict__ bcat) {
  int i = blockIdx.x * 256 + threadIdx.x;     // 0..3071
  float v = (i < 1024) ? bq[i] : (i < 2048) ? bk[i - 1024] : bv[i - 2048];
  bcat[i] = v;
}

// ---------------------------------------------------------------- top-k
__global__ __launch_bounds__(256) void k_topk(const float* __restrict__ mask,
                                              int* __restrict__ topidx) {
  __shared__ float vals[L_];
  __shared__ float rv[256];
  __shared__ int   ri[256];
  const int b = blockIdx.x, tid = threadIdx.x;
  for (int i = tid; i < L_; i += 256) vals[i] = mask[b * L_ + i];
  __syncthreads();
  for (int sel = 0; sel < TOPK_; ++sel) {
    float best = -3.0e38f; int bi = 0;
    for (int i = tid; i < L_; i += 256) {
      float v = vals[i];
      if (v > best) { best = v; bi = i; }
    }
    rv[tid] = best; ri[tid] = bi;
    __syncthreads();
    for (int s = 128; s > 0; s >>= 1) {
      if (tid < s) {
        if (rv[tid + s] > rv[tid] ||
            (rv[tid + s] == rv[tid] && ri[tid + s] < ri[tid])) {
          rv[tid] = rv[tid + s]; ri[tid] = ri[tid + s];
        }
      }
      __syncthreads();
    }
    if (tid == 0) { topidx[b * TOPK_ + sel] = ri[0]; vals[ri[0]] = -3.0e38f; }
    __syncthreads();
  }
}

// ---------------------------------------------------------------- WMMA GEMM
// C[M,N] = A[M,K](bf16) * Bt[N,K]^T(bf16) + bias[N].
// Block tile 128x64 (8 waves in 4x2), wave tile 32x32 (2x2 WMMA tiles),
// double-buffered LDS staging of A(128x32) and B(64x32) per k-step.
#define ASTR 40   // padded LDS row stride (bf16 elems)

template <bool F32OUT>
__global__ __launch_bounds__(256) void k_gemm_bias(const unsigned short* __restrict__ A,
                                                   const unsigned short* __restrict__ Bt,
                                                   const float* __restrict__ bias,
                                                   void* __restrict__ outp,
                                                   int N, int K) {
  __shared__ __align__(16) unsigned short As[2][128 * ASTR];
  __shared__ __align__(16) unsigned short Bs[2][64 * ASTR];

  const int tid = threadIdx.x;
  const int w = tid >> 5, lane = tid & 31;
  const int nl = lane & 15, kh = lane >> 4;
  const int wr = w >> 1, wc = w & 1;                 // 4x2 wave grid
  const int bm = blockIdx.x * 128, bn = blockIdx.y * 64;

  // global staging addresses (each thread: 2 A-chunks + 1 B-chunk of 8 bf16)
  const int srow = tid >> 2, sq = (tid & 3) * 8;
  const unsigned short* aG0 = A + (size_t)(bm + srow) * K + sq;
  const unsigned short* aG1 = A + (size_t)(bm + 64 + srow) * K + sq;
  const unsigned short* bG  = Bt + (size_t)(bn + srow) * K + sq;
  const int sOff = srow * ASTR + sq;

  const int steps = K >> 5;
  uint4 ra0, ra1, rb;

  // prologue: stage step 0, prefetch step 1
  ra0 = *(const uint4*)(aG0);
  ra1 = *(const uint4*)(aG1);
  rb  = *(const uint4*)(bG);
  *(uint4*)(&As[0][sOff])            = ra0;
  *(uint4*)(&As[0][64 * ASTR + sOff]) = ra1;
  *(uint4*)(&Bs[0][sOff])            = rb;
  if (steps > 1) {
    ra0 = *(const uint4*)(aG0 + 32);
    ra1 = *(const uint4*)(aG1 + 32);
    rb  = *(const uint4*)(bG + 32);
  }
  __syncthreads();

  v8f acc[2][2];
#pragma unroll
  for (int mi = 0; mi < 2; ++mi)
#pragma unroll
    for (int ni = 0; ni < 2; ++ni) acc[mi][ni] = (v8f){0, 0, 0, 0, 0, 0, 0, 0};

  for (int s = 0; s < steps; ++s) {
    const int cur = s & 1;
    const unsigned short* as = As[cur];
    const unsigned short* bs = Bs[cur];

    FragBF af[2], bfr[2];
#pragma unroll
    for (int mi = 0; mi < 2; ++mi) {
      const unsigned short* ap = &as[(wr * 32 + mi * 16 + nl) * ASTR];
      af[mi].q[0] = *(const uint4*)(ap + kh * 8);
      af[mi].q[1] = *(const uint4*)(ap + 16 + kh * 8);
    }
#pragma unroll
    for (int ni = 0; ni < 2; ++ni) {
      const unsigned short* bp = &bs[(wc * 32 + ni * 16 + nl) * ASTR];
      bfr[ni].q[0] = *(const uint4*)(bp + kh * 8);
      bfr[ni].q[1] = *(const uint4*)(bp + 16 + kh * 8);
    }
#pragma unroll
    for (int mi = 0; mi < 2; ++mi)
#pragma unroll
      for (int ni = 0; ni < 2; ++ni)
        acc[mi][ni] = wmma_bf16(af[mi].v, bfr[ni].v, acc[mi][ni]);

    if (s + 1 < steps) {
      const int nxt = (s + 1) & 1;
      *(uint4*)(&As[nxt][sOff])             = ra0;
      *(uint4*)(&As[nxt][64 * ASTR + sOff]) = ra1;
      *(uint4*)(&Bs[nxt][sOff])             = rb;
      if (s + 2 < steps) {
        const int kk = (s + 2) * 32;
        ra0 = *(const uint4*)(aG0 + kk);
        ra1 = *(const uint4*)(aG1 + kk);
        rb  = *(const uint4*)(bG + kk);
      }
      __syncthreads();
    }
  }

  // epilogue: bias + store
#pragma unroll
  for (int ni = 0; ni < 2; ++ni) {
    const int col = bn + wc * 32 + ni * 16 + nl;
    const float bv = bias[col];
#pragma unroll
    for (int mi = 0; mi < 2; ++mi) {
      const int row0 = bm + wr * 32 + mi * 16 + kh * 8;
      if (F32OUT) {
        float* out = (float*)outp;
#pragma unroll
        for (int i = 0; i < 8; ++i)
          out[(size_t)(row0 + i) * N + col] = acc[mi][ni][i] + bv;
      } else {
        unsigned short* out = (unsigned short*)outp;
#pragma unroll
        for (int i = 0; i < 8; ++i)
          out[(size_t)(row0 + i) * N + col] = f2bf(acc[mi][ni][i] + bv);
      }
    }
  }
}

// ---------------------------------------------------------------- attention
// grid = (NC, H, B), 256 threads (8 waves). Wave w handles queries
// [c*128 + 16w, c*128 + 16w + 16). Keys: 256 local-window + 64 global top-k.
// Q/K/V come from the fused QKV buffer with row stride QKVS.
#define KSTR 72    // padded K row stride (bf16 elems)
#define VSTR 328   // padded Vt / P row stride (bf16 elems)

__global__ __launch_bounds__(256) void k_attention(const unsigned short* __restrict__ qkv,
                                                   const int* __restrict__ topidx,
                                                   unsigned short* __restrict__ Ab) {
  // shA: phase 1-2 = K tiles [320][KSTR]; phase 4-5 = P [8 waves][16][VSTR]
  __shared__ __align__(16) unsigned short shA[8 * 16 * VSTR];   // 84 KB
  __shared__ __align__(16) unsigned short shV[DH * VSTR];       // 42 KB

  const int c = blockIdx.x, h = blockIdx.y, b = blockIdx.z;
  const int tid = threadIdx.x;
  const int w = tid >> 5, lane = tid & 31;
  const int nl = lane & 15, kh = lane >> 4;

  // ---- phase 1: stage K (row-major) and V (transposed) into LDS ----------
  for (int it = 0; it < 10; ++it) {                 // 320 rows x 8 quads
    int ch = it * 256 + tid;
    int j = ch >> 3, qd = ch & 7;
    int pos;
    if (j < LW_) {
      int p = c * W_ - W_ / 2 + j;
      pos = min(max(p, 0), L_ - 1);
    } else {
      pos = topidx[b * TOPK_ + (j - LW_)];
    }
    size_t base = (size_t)(b * L_ + pos) * QKVS + h * DH + qd * 8;
    uint4 kq = *(const uint4*)(qkv + base + KOFF_);
    *(uint4*)(&shA[j * KSTR + qd * 8]) = kq;
    union { uint4 q; unsigned short s[8]; } vq;
    vq.q = *(const uint4*)(qkv + base + VOFF_);
#pragma unroll
    for (int e = 0; e < 8; ++e) shV[(qd * 8 + e) * VSTR + j] = vq.s[e];
  }
  __syncthreads();

  // ---- phase 2: S = Q K^T via WMMA (20 key-tiles x 2 k-chunks) ------------
  FragBF aq[2];
  {
    const unsigned short* qrow =
        qkv + (size_t)(b * L_ + c * W_ + w * 16 + nl) * QKVS + h * DH;
    aq[0].q[0] = *(const uint4*)(qrow + kh * 8);
    aq[0].q[1] = *(const uint4*)(qrow + 16 + kh * 8);
    aq[1].q[0] = *(const uint4*)(qrow + 32 + kh * 8);
    aq[1].q[1] = *(const uint4*)(qrow + 48 + kh * 8);
  }

  v8f sacc[20];
#pragma unroll
  for (int kt = 0; kt < 20; ++kt) {
    v8f acc = {0, 0, 0, 0, 0, 0, 0, 0};
#pragma unroll
    for (int kc = 0; kc < 2; ++kc) {
      FragBF bk;
      const unsigned short* kp = &shA[(kt * 16 + nl) * KSTR + kc * 32 + kh * 8];
      bk.q[0] = *(const uint4*)(kp);
      bk.q[1] = *(const uint4*)(kp + 16);
      acc = wmma_bf16(aq[kc].v, bk.v, acc);
    }
    sacc[kt] = acc;
  }

  // ---- phase 3: mask + softmax (rows live across 16 lanes) ----------------
  int tg[4];
#pragma unroll
  for (int t = 0; t < 4; ++t) tg[t] = topidx[t * 16 + nl];   // batch-0 indices

  const int qbase = c * W_ + w * 16 + kh * 8;
  float mx[8];
#pragma unroll
  for (int i = 0; i < 8; ++i) mx[i] = -3.0e38f;

#pragma unroll
  for (int kt = 0; kt < 20; ++kt) {
    const int j = kt * 16 + nl;
#pragma unroll
    for (int i = 0; i < 8; ++i) {
      float s = sacc[kt][i] * SCALE_;
      const int qp = qbase + i;
      bool msk;
      if (j < LW_) {
        int p = c * W_ - W_ / 2 + j;
        msk = (qp < p) || (p < 0) || (p >= L_);
      } else {
        msk = qp < tg[(j - LW_) >> 4];
      }
      s = msk ? -1.0e30f : s;
      sacc[kt][i] = s;
      mx[i] = fmaxf(mx[i], s);
    }
  }
#pragma unroll
  for (int i = 0; i < 8; ++i) {
#pragma unroll
    for (int o = 1; o < 16; o <<= 1) mx[i] = fmaxf(mx[i], __shfl_xor(mx[i], o, 16));
  }
  float sm[8];
#pragma unroll
  for (int i = 0; i < 8; ++i) sm[i] = 0.0f;
#pragma unroll
  for (int kt = 0; kt < 20; ++kt) {
#pragma unroll
    for (int i = 0; i < 8; ++i) {
      float p = __expf(sacc[kt][i] - mx[i]);
      sacc[kt][i] = p;
      sm[i] += p;
    }
  }
#pragma unroll
  for (int i = 0; i < 8; ++i) {
#pragma unroll
    for (int o = 1; o < 16; o <<= 1) sm[i] += __shfl_xor(sm[i], o, 16);
    sm[i] = 1.0f / sm[i];
  }

  // ---- phase 4: write P (bf16) into shA, reusing the K region -------------
  __syncthreads();   // all waves done reading K from shA
  const int pbase = w * 16 * VSTR;
#pragma unroll
  for (int kt = 0; kt < 20; ++kt) {
#pragma unroll
    for (int i = 0; i < 8; ++i) {
      shA[pbase + (kh * 8 + i) * VSTR + kt * 16 + nl] = f2bf(sacc[kt][i] * sm[i]);
    }
  }
  __syncthreads();

  // ---- phase 5: O = P V via WMMA (4 d-tiles x 10 key-chunks) --------------
#pragma unroll
  for (int ot = 0; ot < 4; ++ot) {
    v8f oc = {0, 0, 0, 0, 0, 0, 0, 0};
#pragma unroll
    for (int kk = 0; kk < 10; ++kk) {
      FragBF ap, bv;
      const unsigned short* pp = &shA[pbase + nl * VSTR + kk * 32];
      ap.q[0] = *(const uint4*)(pp + kh * 8);
      ap.q[1] = *(const uint4*)(pp + 16 + kh * 8);
      const unsigned short* vp = &shV[(ot * 16 + nl) * VSTR + kk * 32 + kh * 8];
      bv.q[0] = *(const uint4*)(vp);
      bv.q[1] = *(const uint4*)(vp + 16);
      oc = wmma_bf16(ap.v, bv.v, oc);
    }
    const int col = h * DH + ot * 16 + nl;
    const size_t rowb = (size_t)(b * L_ + c * W_ + w * 16 + kh * 8) * D_;
#pragma unroll
    for (int i = 0; i < 8; ++i) Ab[rowb + (size_t)i * D_ + col] = f2bf(oc[i]);
  }
}

// ---------------------------------------------------------------- launcher
extern "C" void kernel_launch(void* const* d_in, const int* in_sizes, int n_in,
                              void* d_out, int out_size, void* d_ws, size_t ws_size,
                              hipStream_t stream) {
  (void)in_sizes; (void)n_in; (void)out_size; (void)ws_size;

  const float* x    = (const float*)d_in[0];
  const float* imp  = (const float*)d_in[1];
  const float* Wq   = (const float*)d_in[2];
  const float* bq   = (const float*)d_in[3];
  const float* Wk   = (const float*)d_in[4];
  const float* bk   = (const float*)d_in[5];
  const float* Wv   = (const float*)d_in[6];
  const float* bv   = (const float*)d_in[7];
  const float* Wo   = (const float*)d_in[8];
  const float* bo   = (const float*)d_in[9];
  float* out = (float*)d_out;

  char* ws = (char*)d_ws;
  const size_t SZ_X   = (size_t)B_ * L_ * D_ * 2;       // 16 MB   bf16 x / attn-out
  const size_t SZ_W   = (size_t)D_ * D_ * 2;            // 2 MB    bf16 weight
  const size_t SZ_QKV = (size_t)B_ * L_ * QKVS * 2;     // 50.3 MB fused QKV

  size_t off = 0;
  unsigned short* xb   = (unsigned short*)(ws + off); off += SZ_X;
  unsigned short* wcat = (unsigned short*)(ws + off); off += 3 * SZ_W;  // [3072][1024]
  unsigned short* woT  = (unsigned short*)(ws + off); off += SZ_W;
  float*          bcat = (float*)(ws + off);          off += QKVS * sizeof(float);
  off = (off + 255) & ~(size_t)255;
  unsigned short* qkvb = (unsigned short*)(ws + off); off += SZ_QKV;
  unsigned short* Ab   = (unsigned short*)(ws + off); off += SZ_X;
  int*          topidx = (int*)(ws + off);

  const int NTOK = B_ * L_;              // 8192
  const int NX   = NTOK * D_;            // 8,388,608

  // 1) conversions / layout prep
  k_f32_to_bf16<<<NX / 256, 256, 0, stream>>>(x, xb, NX);
  k_transpose_bf16<<<(D_ * D_) / 256, 256, 0, stream>>>(Wq, wcat);
  k_transpose_bf16<<<(D_ * D_) / 256, 256, 0, stream>>>(Wk, wcat + (size_t)D_ * D_);
  k_transpose_bf16<<<(D_ * D_) / 256, 256, 0, stream>>>(Wv, wcat + (size_t)2 * D_ * D_);
  k_transpose_bf16<<<(D_ * D_) / 256, 256, 0, stream>>>(Wo, woT);
  k_concat_bias<<<QKVS / 256, 256, 0, stream>>>(bq, bk, bv, bcat);

  // 2) top-k indices per batch
  k_topk<<<B_, 256, 0, stream>>>(imp, topidx);

  // 3) fused QKV projection: [8192,1024] x [1024,3072] -> [8192,3072] bf16
  dim3 gq(NTOK / 128, QKVS / 64);        // (64, 48)
  k_gemm_bias<false><<<gq, 256, 0, stream>>>(xb, wcat, bcat, qkvb, QKVS, D_);

  // 4) block-sparse attention
  dim3 ag(NC_, H_, B_);
  k_attention<<<ag, 256, 0, stream>>>(qkvb, topidx, Ab);

  // 5) output projection (fp32 out): [8192,1024] x [1024,1024] -> d_out
  dim3 go(NTOK / 128, D_ / 64);          // (64, 16)
  k_gemm_bias<true><<<go, 256, 0, stream>>>(Ab, woT, bo, out, D_, D_);
}
